// PixelDecoder_31799937860128
// MI455X (gfx1250) — compile-verified, matrix operand order checked by reading, not run on hardware
//
#include <hip/hip_runtime.h>
#include <hip/hip_bf16.h>
#include <math.h>

typedef __bf16 bf16;
typedef __attribute__((ext_vector_type(16))) __bf16 v16bf;
typedef __attribute__((ext_vector_type(8)))  __bf16 v8bf;
typedef __attribute__((ext_vector_type(8)))  float  v8f;

#define B_SZ    8
#define NP_SEQ  784
#define ED_IN   1024
#define D_MODEL 512
#define H_HEADS 8
#define DH_HEAD 64
#define NPATCH  1568
#define LAYERS  4
#define FF_DIM  2048
#define OUT_D   1536
#define M_X     (B_SZ * NP_SEQ)   /* 6272 */

// ---------------------------------------------------------------------------
// 16-bf16 operand fragment for v_wmma_f32_16x16x32_bf16.
// p -> (row_base + k0 + koffb); elements 0..7 = k offsets 0..7,
// elements 8..15 = k offsets 16..23 (lanes>=16 add +8 via koffb).
// Two 16-byte vector loads, no conversion ALU.
// ---------------------------------------------------------------------------
__device__ __forceinline__ v16bf load_frag_bf16(const bf16* __restrict__ p) {
  const v8bf lo = *(const v8bf*)p;
  const v8bf hi = *(const v8bf*)(p + 16);
  v16bf f;
#pragma unroll
  for (int i = 0; i < 8; ++i) { f[i] = lo[i]; f[i + 8] = hi[i]; }
  return f;
}

__device__ __forceinline__ float gelu_exact(float x) {
  return 0.5f * x * (1.0f + erff(x * 0.70710678118654752f));
}

// ---------------------------------------------------------------------------
// C = act(A[M,K] @ W[N,K]^T + bias)   A,W bf16; C f32 (optional); Cb bf16
// (optional). One wave computes a 32x64 tile: 2 A-frags x 4 B-frags ->
// 8 WMMAs per 32-deep K step. Requires M%32==0, N%64==0, K%32==0.
// ---------------------------------------------------------------------------
__global__ void gemm_wmma_kernel(const bf16* __restrict__ A,
                                 const bf16* __restrict__ W,
                                 const float* __restrict__ bias,
                                 float* __restrict__ C,
                                 bf16* __restrict__ Cb,
                                 int M, int N, int K, int act) {
  const int lane = threadIdx.x & 31;
  const int wave = blockIdx.x * (blockDim.x >> 5) + (threadIdx.x >> 5);
  const int ntn  = N >> 6;
  const int tm   = wave / ntn;
  const int m0   = tm << 5;
  if (m0 >= M) return;
  const int n0    = (wave - tm * ntn) << 6;
  const int koffb = (lane & 16) ? 8 : 0;
  const int ln15  = lane & 15;

  const bf16* a0p = A + (size_t)(m0 + ln15) * K + koffb;
  const bf16* a1p = a0p + (size_t)16 * K;
  const bf16* w0  = W + (size_t)(n0 + ln15) * K + koffb;
  const bf16* w1  = w0 + (size_t)16 * K;
  const bf16* w2  = w0 + (size_t)32 * K;
  const bf16* w3  = w0 + (size_t)48 * K;

  v8f c00 = {}, c01 = {}, c02 = {}, c03 = {};
  v8f c10 = {}, c11 = {}, c12 = {}, c13 = {};
  for (int k0 = 0; k0 < K; k0 += 32) {
    __builtin_prefetch(a0p + k0 + 128, 0, 0);   // global_prefetch_b8 (stream A)
    const v16bf a0 = load_frag_bf16(a0p + k0);
    const v16bf a1 = load_frag_bf16(a1p + k0);
    const v16bf b0 = load_frag_bf16(w0 + k0);
    const v16bf b1 = load_frag_bf16(w1 + k0);
    const v16bf b2 = load_frag_bf16(w2 + k0);
    const v16bf b3 = load_frag_bf16(w3 + k0);
    c00 = __builtin_amdgcn_wmma_f32_16x16x32_bf16(false, a0, false, b0, (short)0, c00, false, false);
    c01 = __builtin_amdgcn_wmma_f32_16x16x32_bf16(false, a0, false, b1, (short)0, c01, false, false);
    c02 = __builtin_amdgcn_wmma_f32_16x16x32_bf16(false, a0, false, b2, (short)0, c02, false, false);
    c03 = __builtin_amdgcn_wmma_f32_16x16x32_bf16(false, a0, false, b3, (short)0, c03, false, false);
    c10 = __builtin_amdgcn_wmma_f32_16x16x32_bf16(false, a1, false, b0, (short)0, c10, false, false);
    c11 = __builtin_amdgcn_wmma_f32_16x16x32_bf16(false, a1, false, b1, (short)0, c11, false, false);
    c12 = __builtin_amdgcn_wmma_f32_16x16x32_bf16(false, a1, false, b2, (short)0, c12, false, false);
    c13 = __builtin_amdgcn_wmma_f32_16x16x32_bf16(false, a1, false, b3, (short)0, c13, false, false);
  }

  const int hi = (lane >> 4) << 3;  // C layout: VGPR r -> row r (+8 for lanes 16..31)
  const float bv0 = bias[n0 + ln15];
  const float bv1 = bias[n0 + 16 + ln15];
  const float bv2 = bias[n0 + 32 + ln15];
  const float bv3 = bias[n0 + 48 + ln15];
#pragma unroll
  for (int r = 0; r < 8; ++r) {
#pragma unroll
    for (int half = 0; half < 2; ++half) {
      const int row = m0 + half * 16 + hi + r;
      float v0 = (half ? c10[r] : c00[r]) + bv0;
      float v1 = (half ? c11[r] : c01[r]) + bv1;
      float v2 = (half ? c12[r] : c02[r]) + bv2;
      float v3 = (half ? c13[r] : c03[r]) + bv3;
      if (act) { v0 = gelu_exact(v0); v1 = gelu_exact(v1); v2 = gelu_exact(v2); v3 = gelu_exact(v3); }
      if (C) {
        float* cr = C + (size_t)row * N + n0 + ln15;
        cr[0] = v0; cr[16] = v1; cr[32] = v2; cr[48] = v3;
      }
      if (Cb) {
        bf16* cb = Cb + (size_t)row * N + n0 + ln15;
        cb[0] = (bf16)v0; cb[16] = (bf16)v1; cb[32] = (bf16)v2; cb[48] = (bf16)v3;
      }
    }
  }
}

// ---------------------------------------------------------------------------
// Flash-style attention, one (batch, head, 16-query tile) per wave.
// Q/K/V/O all bf16 (head cols ch..ch+63 within row stride). Online softmax
// over 32-key steps; P tile bounced through LDS into an A operand.
// ---------------------------------------------------------------------------
__global__ void attn_wmma_kernel(const bf16* __restrict__ Q, int qstride,
                                 const bf16* __restrict__ Km, int kstride,
                                 const bf16* __restrict__ V,  int vstride,
                                 bf16* __restrict__ O, int ostride,
                                 int Sq, int Skv, int kv_brows) {
  __shared__ bf16 Plds[16][32];
  const int lane = threadIdx.x & 31;
  const int nqt  = Sq >> 4;
  int idx = blockIdx.x;
  const int qt = idx % nqt; idx /= nqt;
  const int h  = idx % H_HEADS; idx /= H_HEADS;
  const int b  = idx;
  const int ch     = h * DH_HEAD;
  const int qrow0  = b * Sq + qt * 16;
  const int kvrow0 = b * kv_brows;
  const int koffb  = (lane & 16) ? 8 : 0;
  const int ln15   = lane & 15;
  const int hi     = (lane >> 4) << 3;

  const bf16* qp = Q + (size_t)(qrow0 + ln15) * qstride + ch + koffb;
  const v16bf qa0 = load_frag_bf16(qp);
  const v16bf qa1 = load_frag_bf16(qp + 32);

  v8f o0 = {}, o1 = {}, o2 = {}, o3 = {};
  float mrun[8], lrun[8];
#pragma unroll
  for (int r = 0; r < 8; ++r) { mrun[r] = -3.0e38f; lrun[r] = 0.0f; }
  const float scale = 0.125f;  // 1/sqrt(64)

  for (int kt = 0; kt < Skv; kt += 32) {
    // ---- scores S[16q x 32k] = Q @ K^T ----------------------------------
    v8f sc0 = {}, sc1 = {};
    const int key0 = kt + ln15, key1 = kt + 16 + ln15;
    const int ck0  = (key0 < Skv) ? key0 : Skv - 1;
    const int ck1  = (key1 < Skv) ? key1 : Skv - 1;
    const bf16* kp0 = Km + (size_t)(kvrow0 + ck0) * kstride + ch + koffb;
    const bf16* kp1 = Km + (size_t)(kvrow0 + ck1) * kstride + ch + koffb;
    v16bf kb;
    kb  = load_frag_bf16(kp0);
    sc0 = __builtin_amdgcn_wmma_f32_16x16x32_bf16(false, qa0, false, kb, (short)0, sc0, false, false);
    kb  = load_frag_bf16(kp0 + 32);
    sc0 = __builtin_amdgcn_wmma_f32_16x16x32_bf16(false, qa1, false, kb, (short)0, sc0, false, false);
    kb  = load_frag_bf16(kp1);
    sc1 = __builtin_amdgcn_wmma_f32_16x16x32_bf16(false, qa0, false, kb, (short)0, sc1, false, false);
    kb  = load_frag_bf16(kp1 + 32);
    sc1 = __builtin_amdgcn_wmma_f32_16x16x32_bf16(false, qa1, false, kb, (short)0, sc1, false, false);

    // ---- online softmax (row = r + hi, keys across 16 lanes) ------------
#pragma unroll
    for (int r = 0; r < 8; ++r) {
      float s0 = (key0 < Skv) ? sc0[r] * scale : -1.0e30f;
      float s1 = (key1 < Skv) ? sc1[r] * scale : -1.0e30f;
      float tm = fmaxf(s0, s1);
#pragma unroll
      for (int off = 1; off < 16; off <<= 1) tm = fmaxf(tm, __shfl_xor(tm, off));
      const float mnew = fmaxf(mrun[r], tm);
      const float sf   = __expf(mrun[r] - mnew);
      const float p0   = __expf(s0 - mnew);
      const float p1   = __expf(s1 - mnew);
      float ts = p0 + p1;
#pragma unroll
      for (int off = 1; off < 16; off <<= 1) ts += __shfl_xor(ts, off);
      lrun[r] = lrun[r] * sf + ts;
      mrun[r] = mnew;
      o0[r] *= sf; o1[r] *= sf; o2[r] *= sf; o3[r] *= sf;
      Plds[r + hi][ln15]      = (bf16)p0;
      Plds[r + hi][16 + ln15] = (bf16)p1;
    }
    __syncthreads();

    // ---- A fragment from LDS P tile (two 16B ds loads) ------------------
    const v16bf pa = load_frag_bf16(&Plds[ln15][koffb]);

    // ---- V fragments: B[k][n] = V[kt+k][ch + n] -------------------------
    int ckk[16];
#pragma unroll
    for (int i = 0; i < 16; ++i) {
      const int kk = kt + ((i >= 8) ? 16 : 0) + koffb + (i & 7);
      ckk[i] = (kk < Skv) ? kk : Skv - 1;
    }
    v16bf vb0, vb1, vb2, vb3;
#pragma unroll
    for (int i = 0; i < 16; ++i) {
      const bf16* vp = V + (size_t)(kvrow0 + ckk[i]) * vstride + ch + ln15;
      vb0[i] = vp[0];
      vb1[i] = vp[16];
      vb2[i] = vp[32];
      vb3[i] = vp[48];
    }
    o0 = __builtin_amdgcn_wmma_f32_16x16x32_bf16(false, pa, false, vb0, (short)0, o0, false, false);
    o1 = __builtin_amdgcn_wmma_f32_16x16x32_bf16(false, pa, false, vb1, (short)0, o1, false, false);
    o2 = __builtin_amdgcn_wmma_f32_16x16x32_bf16(false, pa, false, vb2, (short)0, o2, false, false);
    o3 = __builtin_amdgcn_wmma_f32_16x16x32_bf16(false, pa, false, vb3, (short)0, o3, false, false);
    __syncthreads();
  }

#pragma unroll
  for (int r = 0; r < 8; ++r) {
    const float inv = 1.0f / lrun[r];
    bf16* op = O + (size_t)(qrow0 + hi + r) * ostride + ch + ln15;
    op[0]  = (bf16)(o0[r] * inv);
    op[16] = (bf16)(o1[r] * inv);
    op[32] = (bf16)(o2[r] * inv);
    op[48] = (bf16)(o3[r] * inv);
  }
}

// ---------------------------------------------------------------------------
// x[row] = LayerNorm(x[row] + h[row]) * w + b ; also emits bf16 copy xb.
// D = 512, 256 threads/row. h may be null (plain LN).
// ---------------------------------------------------------------------------
__device__ __forceinline__ float block_sum256(float v, float* sm) {
#pragma unroll
  for (int off = 16; off > 0; off >>= 1) v += __shfl_xor(v, off);
  const int wid = threadIdx.x >> 5;
  if ((threadIdx.x & 31) == 0) sm[wid] = v;
  __syncthreads();
  float r = (threadIdx.x < 8) ? sm[threadIdx.x] : 0.0f;
#pragma unroll
  for (int off = 4; off > 0; off >>= 1) r += __shfl_xor(r, off);
  if (threadIdx.x == 0) sm[8] = r;
  __syncthreads();
  const float out = sm[8];
  __syncthreads();
  return out;
}

__global__ void add_ln_kernel(float* __restrict__ x, const float* __restrict__ h,
                              const float* __restrict__ w, const float* __restrict__ b,
                              bf16* __restrict__ xb) {
  __shared__ float sm[9];
  const int row = blockIdx.x;
  const int tid = threadIdx.x;
  float* xr = x + (size_t)row * D_MODEL;
  float v0 = xr[tid], v1 = xr[tid + 256];
  if (h) {
    const float* hr = h + (size_t)row * D_MODEL;
    v0 += hr[tid]; v1 += hr[tid + 256];
  }
  const float mean = block_sum256(v0 + v1, sm) * (1.0f / 512.0f);
  const float d0 = v0 - mean, d1 = v1 - mean;
  const float var  = block_sum256(d0 * d0 + d1 * d1, sm) * (1.0f / 512.0f);
  const float rstd = rsqrtf(var + 1e-5f);
  const float y0 = d0 * rstd * w[tid]       + b[tid];
  const float y1 = d1 * rstd * w[tid + 256] + b[tid + 256];
  xr[tid]       = y0;
  xr[tid + 256] = y1;
  bf16* xbr = xb + (size_t)row * D_MODEL;
  xbr[tid]       = (bf16)y0;
  xbr[tid + 256] = (bf16)y1;
}

// ---------------------------------------------------------------------------
__global__ void pos_embed_kernel(bf16* __restrict__ memb) {
  const int idx = blockIdx.x * blockDim.x + threadIdx.x;
  if (idx >= NPATCH * 256) return;
  const int p = idx >> 8, i = idx & 255;
  const float div = __expf((float)(2 * i) * (-9.210340371976184f / 512.0f));
  const float ang = (float)p * div;
  memb[(size_t)p * D_MODEL + 2 * i]     = (bf16)sinf(ang);
  memb[(size_t)p * D_MODEL + 2 * i + 1] = (bf16)cosf(ang);
}

__global__ void f32_to_bf16_kernel(const float* __restrict__ s, bf16* __restrict__ d, long n) {
  const long i = ((long)blockIdx.x * blockDim.x + threadIdx.x) * 4;
  if (i >= n) return;  // all sizes are multiples of 4
  d[i]     = (bf16)s[i];
  d[i + 1] = (bf16)s[i + 1];
  d[i + 2] = (bf16)s[i + 2];
  d[i + 3] = (bf16)s[i + 3];
}

__global__ void zero_kernel(float* __restrict__ p, long n) {
  for (long i = (long)blockIdx.x * blockDim.x + threadIdx.x; i < n;
       i += (long)gridDim.x * blockDim.x)
    p[i] = 0.0f;
}

// Scatter pred [B, NP, 2, 3, 16, 16] into frames [B, 16, 3, 224, 224]
__global__ void scatter_kernel(const float* __restrict__ pred,
                               const int* __restrict__ mask,
                               float* __restrict__ out) {
  const long e = (long)blockIdx.x * blockDim.x + threadIdx.x;
  if (e >= (long)M_X * OUT_D) return;
  const int  j  = (int)(e % OUT_D);
  const long bp = e / OUT_D;
  const int p = (int)(bp % NP_SEQ);
  const int b = (int)(bp / NP_SEQ);
  const int f  = j / 768;
  const int c  = (j / 256) % 3;
  const int pi = (j / 16) % 16;
  const int pj = j % 16;
  const int m  = mask[b * NP_SEQ + p];
  const int t  = m / 196, s = m % 196, hh = s / 14, ww = s % 14;
  const int frame = t * 2 + f;
  const size_t dst = ((((size_t)b * 16 + frame) * 3 + c) * 224 + (hh * 16 + pi)) * 224
                     + (ww * 16 + pj);
  out[dst] = pred[e];
}

// ---------------------------------------------------------------------------
extern "C" void kernel_launch(void* const* d_in, const int* in_sizes, int n_in,
                              void* d_out, int out_size, void* d_ws, size_t ws_size,
                              hipStream_t stream) {
  const float* z         = (const float*)d_in[0];
  const int*   mask_pred = (const int*)  d_in[1];
  const float* embed_w   = (const float*)d_in[2];
  const float* embed_b   = (const float*)d_in[3];
  const float* sa_qkv_w  = (const float*)d_in[4];
  const float* sa_qkv_b  = (const float*)d_in[5];
  const float* sa_out_w  = (const float*)d_in[6];
  const float* sa_out_b  = (const float*)d_in[7];
  const float* ca_qkv_w  = (const float*)d_in[8];
  const float* ca_qkv_b  = (const float*)d_in[9];
  const float* ca_out_w  = (const float*)d_in[10];
  const float* ca_out_b  = (const float*)d_in[11];
  const float* ff1_w     = (const float*)d_in[12];
  const float* ff1_b     = (const float*)d_in[13];
  const float* ff2_w     = (const float*)d_in[14];
  const float* ff2_b     = (const float*)d_in[15];
  const float* ln1_w     = (const float*)d_in[16];
  const float* ln1_b     = (const float*)d_in[17];
  const float* ln2_w     = (const float*)d_in[18];
  const float* ln2_b     = (const float*)d_in[19];
  const float* ln3_w     = (const float*)d_in[20];
  const float* ln3_b     = (const float*)d_in[21];
  const float* norm_w    = (const float*)d_in[22];
  const float* norm_b    = (const float*)d_in[23];
  const float* pred_w    = (const float*)d_in[24];
  const float* pred_b    = (const float*)d_in[25];
  float* out = (float*)d_out;

  // ---- workspace: f32 region then bf16 region ---------------------------
  float* wsf = (float*)d_ws;
  float* x     = wsf; wsf += (size_t)M_X * D_MODEL;
  float* hb    = wsf; wsf += (size_t)M_X * D_MODEL;
  float* predb = wsf; wsf += (size_t)M_X * OUT_D;

  bf16* wsb = (bf16*)wsf;
  bf16* zb    = wsb; wsb += (size_t)M_X * ED_IN;
  bf16* xb    = wsb; wsb += (size_t)M_X * D_MODEL;
  bf16* memb  = wsb; wsb += (size_t)NPATCH * D_MODEL;
  bf16* qkvb  = wsb; wsb += (size_t)M_X * 3 * D_MODEL;
  bf16* kvb   = wsb; wsb += (size_t)NPATCH * 2 * D_MODEL;
  bf16* qbb   = wsb; wsb += (size_t)M_X * D_MODEL;
  bf16* aob   = wsb; wsb += (size_t)M_X * D_MODEL;
  bf16* ffbb  = wsb; wsb += (size_t)M_X * FF_DIM;
  bf16* w_embed = wsb; wsb += (size_t)D_MODEL * ED_IN;
  bf16* w_saqkv = wsb; wsb += (size_t)LAYERS * 3 * D_MODEL * D_MODEL;
  bf16* w_saout = wsb; wsb += (size_t)LAYERS * D_MODEL * D_MODEL;
  bf16* w_caqkv = wsb; wsb += (size_t)LAYERS * 3 * D_MODEL * D_MODEL;
  bf16* w_caout = wsb; wsb += (size_t)LAYERS * D_MODEL * D_MODEL;
  bf16* w_ff1   = wsb; wsb += (size_t)LAYERS * FF_DIM * D_MODEL;
  bf16* w_ff2   = wsb; wsb += (size_t)LAYERS * D_MODEL * FF_DIM;
  bf16* w_pred  = wsb; wsb += (size_t)OUT_D * D_MODEL;

  auto cvt = [&](const float* s, bf16* d, long n) {
    f32_to_bf16_kernel<<<(int)((n / 4 + 255) / 256), 256, 0, stream>>>(s, d, n);
  };
  auto gemm = [&](const bf16* A, const bf16* W, const float* bias,
                  float* C, bf16* Cb, int M, int N, int K, int act) {
    const int waves  = (M / 32) * (N / 64);
    const int blocks = (waves + 7) / 8;
    gemm_wmma_kernel<<<blocks, 256, 0, stream>>>(A, W, bias, C, Cb, M, N, K, act);
  };
  auto attn = [&](const bf16* Q, int qs, const bf16* Kp, int ks,
                  const bf16* Vp, int vs, bf16* O, int os,
                  int Sq, int Skv, int kvb_rows) {
    attn_wmma_kernel<<<B_SZ * H_HEADS * (Sq / 16), 32, 0, stream>>>(
        Q, qs, Kp, ks, Vp, vs, O, os, Sq, Skv, kvb_rows);
  };
  auto add_ln = [&](float* xx, const float* hh, const float* w, const float* b) {
    add_ln_kernel<<<M_X, 256, 0, stream>>>(xx, hh, w, b, xb);
  };

  // ---- one-time bf16 conversions ---------------------------------------
  cvt(z, zb, (long)M_X * ED_IN);
  cvt(embed_w, w_embed, (long)D_MODEL * ED_IN);
  cvt(sa_qkv_w, w_saqkv, (long)LAYERS * 3 * D_MODEL * D_MODEL);
  cvt(sa_out_w, w_saout, (long)LAYERS * D_MODEL * D_MODEL);
  cvt(ca_qkv_w, w_caqkv, (long)LAYERS * 3 * D_MODEL * D_MODEL);
  cvt(ca_out_w, w_caout, (long)LAYERS * D_MODEL * D_MODEL);
  cvt(ff1_w, w_ff1, (long)LAYERS * FF_DIM * D_MODEL);
  cvt(ff2_w, w_ff2, (long)LAYERS * D_MODEL * FF_DIM);
  cvt(pred_w, w_pred, (long)OUT_D * D_MODEL);
  pos_embed_kernel<<<(NPATCH * 256 + 255) / 256, 256, 0, stream>>>(memb);

  // ---- embed: x (f32 residual) + xb (bf16 for next GEMM) ---------------
  gemm(zb, w_embed, embed_b, x, xb, M_X, D_MODEL, ED_IN, 0);

  for (int l = 0; l < LAYERS; ++l) {
    const size_t wq = (size_t)l * 3 * D_MODEL * D_MODEL;
    const size_t bq = (size_t)l * 3 * D_MODEL;
    const size_t wo = (size_t)l * D_MODEL * D_MODEL;
    const size_t bo = (size_t)l * D_MODEL;
    const size_t wl = (size_t)l * D_MODEL;

    // ---- self-attention --------------------------------------------------
    gemm(xb, w_saqkv + wq, sa_qkv_b + bq, nullptr, qkvb, M_X, 3 * D_MODEL, D_MODEL, 0);
    attn(qkvb, 3 * D_MODEL, qkvb + D_MODEL, 3 * D_MODEL, qkvb + 2 * D_MODEL, 3 * D_MODEL,
         aob, D_MODEL, NP_SEQ, NP_SEQ, NP_SEQ);
    gemm(aob, w_saout + wo, sa_out_b + bo, hb, nullptr, M_X, D_MODEL, D_MODEL, 0);
    add_ln(x, hb, ln1_w + wl, ln1_b + wl);

    // ---- cross-attention (K/V from shared pos-embed memory) -------------
    gemm(xb, w_caqkv + wq, ca_qkv_b + bq, nullptr, qbb, M_X, D_MODEL, D_MODEL, 0);
    gemm(memb, w_caqkv + wq + (size_t)D_MODEL * D_MODEL, ca_qkv_b + bq + D_MODEL,
         nullptr, kvb, NPATCH, 2 * D_MODEL, D_MODEL, 0);
    attn(qbb, D_MODEL, kvb, 2 * D_MODEL, kvb + D_MODEL, 2 * D_MODEL,
         aob, D_MODEL, NP_SEQ, NPATCH, 0);
    gemm(aob, w_caout + wo, ca_out_b + bo, hb, nullptr, M_X, D_MODEL, D_MODEL, 0);
    add_ln(x, hb, ln2_w + wl, ln2_b + wl);

    // ---- feed-forward (GELU fused) --------------------------------------
    gemm(xb, w_ff1 + (size_t)l * FF_DIM * D_MODEL, ff1_b + (size_t)l * FF_DIM,
         nullptr, ffbb, M_X, FF_DIM, D_MODEL, 1);
    gemm(ffbb, w_ff2 + (size_t)l * D_MODEL * FF_DIM, ff2_b + bo,
         hb, nullptr, M_X, D_MODEL, FF_DIM, 0);
    add_ln(x, hb, ln3_w + wl, ln3_b + wl);
  }

  // ---- final norm + prediction head ------------------------------------
  add_ln(x, nullptr, norm_w, norm_b);
  gemm(xb, w_pred, pred_b, predb, nullptr, M_X, OUT_D, D_MODEL, 0);

  // ---- reconstruct frames ----------------------------------------------
  const long out_n = (long)B_SZ * 16 * 3 * 224 * 224;
  zero_kernel<<<4096, 256, 0, stream>>>(out, out_n);
  const long pred_n = (long)M_X * OUT_D;
  scatter_kernel<<<(int)((pred_n + 255) / 256), 256, 0, stream>>>(predb, mask_pred, out);
}
